// vit_TransformerBlock_30047591203054
// MI455X (gfx1250) — compile-verified, matrix operand order checked by reading, not run on hardware
//
#include <hip/hip_runtime.h>
#include <hip/hip_bf16.h>
#include <cstdint>

// ---------------------------------------------------------------------------
// Types for CDNA5 WMMA (gfx1250, wave32)
// ---------------------------------------------------------------------------
typedef __bf16 bf16_t;
typedef __attribute__((ext_vector_type(16))) __bf16 v16bf;
typedef __attribute__((ext_vector_type(8)))  __bf16 v8bf;
typedef __attribute__((ext_vector_type(8)))  float  v8f;

#define D_MODEL 768
#define SEQ     1024
#define BATCH   16
#define NHEAD   8
#define HDIM    96
#define FFDIM   3072
#define TOKENS  (BATCH * SEQ)   // 16384

__device__ __forceinline__ v16bf mk16(v8bf lo, v8bf hi) {
    v16bf r;
#pragma unroll
    for (int i = 0; i < 8; ++i) { r[i] = lo[i]; r[i + 8] = hi[i]; }
    return r;
}

__device__ __forceinline__ v8f wmma_bf16(v16bf a, v16bf b, v8f c) {
    // D(f32 16x16) = A(bf16 16x32) * B(bf16 32x16) + C
    return __builtin_amdgcn_wmma_f32_16x16x32_bf16(
        /*neg_a=*/false, a, /*neg_b=*/false, b,
        /*c_mod=*/(short)0, c, /*reuse_a=*/false, /*reuse_b=*/false);
}

// ---------------------------------------------------------------------------
// Weight transpose + f32 -> bf16 downcast:  in[K][N] -> out[N][K]
// ---------------------------------------------------------------------------
__global__ void transpose_to_bf16(const float* __restrict__ in,
                                  bf16_t* __restrict__ out, int K, int N) {
    __shared__ bf16_t tile[32][33];
    const int n0 = blockIdx.x * 32;
    const int k0 = blockIdx.y * 32;
    const int tx = threadIdx.x;      // 0..31
    const int ty = threadIdx.y;      // 0..7
#pragma unroll
    for (int i = 0; i < 32; i += 8)
        tile[ty + i][tx] = (bf16_t)in[(size_t)(k0 + ty + i) * N + (n0 + tx)];
    __syncthreads();
#pragma unroll
    for (int i = 0; i < 32; i += 8)
        out[(size_t)(n0 + ty + i) * K + (k0 + tx)] = tile[tx][ty + i];
}

// ---------------------------------------------------------------------------
// LayerNorm (population variance, eps=1e-5): f32 in -> bf16 out. 1 block/row.
// ---------------------------------------------------------------------------
__global__ void layernorm_bf16(const float* __restrict__ in,
                               const float* __restrict__ gamma,
                               const float* __restrict__ beta,
                               bf16_t* __restrict__ out) {
    const int row = blockIdx.x;
    const float* x = in + (size_t)row * D_MODEL;
    __shared__ float red[256];

    float s = 0.f, s2 = 0.f;
#pragma unroll
    for (int i = threadIdx.x; i < D_MODEL; i += 256) {
        float v = x[i]; s += v; s2 += v * v;
    }
    red[threadIdx.x] = s;  __syncthreads();
    for (int o = 128; o > 0; o >>= 1) {
        if (threadIdx.x < o) red[threadIdx.x] += red[threadIdx.x + o];
        __syncthreads();
    }
    const float mean = red[0] * (1.0f / D_MODEL);
    __syncthreads();
    red[threadIdx.x] = s2; __syncthreads();
    for (int o = 128; o > 0; o >>= 1) {
        if (threadIdx.x < o) red[threadIdx.x] += red[threadIdx.x + o];
        __syncthreads();
    }
    const float var  = red[0] * (1.0f / D_MODEL) - mean * mean;
    const float rstd = rsqrtf(var + 1e-5f);
#pragma unroll
    for (int i = threadIdx.x; i < D_MODEL; i += 256) {
        float v = (x[i] - mean) * rstd * gamma[i] + beta[i];
        out[(size_t)row * D_MODEL + i] = (bf16_t)v;
    }
}

// ---------------------------------------------------------------------------
// Generic bf16 WMMA GEMM:  C[M x N] = A[M x K] * WT[N x K]^T + bias  (+epilogue)
// Wave tile = 16(M) x 64(N) (4 accumulators); block = 8 waves = 128 x 64.
// MODE 0: bias, store bf16 to q/k layout  [B,H,N,HD]
// MODE 1: bias, store bf16 to v-transposed layout [B,H,HD,N]
// MODE 2: bias + residual(f32), store f32 row-major
// MODE 3: bias + exact GELU, store bf16 row-major
// ---------------------------------------------------------------------------
template <int MODE>
__global__ void gemm_wmma(const bf16_t* __restrict__ A,
                          const bf16_t* __restrict__ WT,
                          const float*  __restrict__ bias,
                          const float*  __restrict__ resid,
                          float*  __restrict__ outF,
                          bf16_t* __restrict__ outB,
                          int M, int N, int K) {
    const int wave = threadIdx.x >> 5;
    const int lane = threadIdx.x & 31;
    const int ln15 = lane & 15;
    const int half = lane >> 4;
    const int m0   = blockIdx.x * 128 + wave * 16;
    const int n0   = blockIdx.y * 64;

    v8f acc[4] = {v8f{}, v8f{}, v8f{}, v8f{}};
    const bf16_t* arow = A + (size_t)(m0 + ln15) * K;

#pragma unroll 2
    for (int k0 = 0; k0 < K; k0 += 32) {
        __builtin_prefetch(arow + k0 + 256, 0, 0);   // global_prefetch_b8
        // phase 1: issue all fragment loads
        v16bf af = mk16(*(const v8bf*)(arow + k0 + 8 * half),
                        *(const v8bf*)(arow + k0 + 16 + 8 * half));
        v16bf bf[4];
#pragma unroll
        for (int t = 0; t < 4; ++t) {
            const bf16_t* brow = WT + (size_t)(n0 + t * 16 + ln15) * K + k0;
            bf[t] = mk16(*(const v8bf*)(brow + 8 * half),
                         *(const v8bf*)(brow + 16 + 8 * half));
        }
        // phase 2: issue all WMMAs
#pragma unroll
        for (int t = 0; t < 4; ++t)
            acc[t] = wmma_bf16(af, bf[t], acc[t]);
    }

#pragma unroll
    for (int t = 0; t < 4; ++t) {
        const int gn = n0 + t * 16 + ln15;
        const float bv = bias[gn];
#pragma unroll
        for (int r = 0; r < 8; ++r) {
            const int gm = m0 + r + 8 * half;
            float v = acc[t][r] + bv;
            if constexpr (MODE == 0) {
                const int b = gm >> 10, tok = gm & 1023;
                const int h = gn / HDIM, hd = gn % HDIM;
                outB[((size_t)(b * NHEAD + h) * SEQ + tok) * HDIM + hd] = (bf16_t)v;
            } else if constexpr (MODE == 1) {
                const int b = gm >> 10, tok = gm & 1023;
                const int h = gn / HDIM, hd = gn % HDIM;
                outB[((size_t)(b * NHEAD + h) * HDIM + hd) * SEQ + tok] = (bf16_t)v;
            } else if constexpr (MODE == 2) {
                const size_t idx = (size_t)gm * N + gn;
                outF[idx] = v + resid[idx];
            } else { // MODE == 3
                const float g = 0.5f * v * (1.0f + erff(v * 0.70710678118f));
                outB[(size_t)gm * N + gn] = (bf16_t)g;
            }
        }
    }
}

// ---------------------------------------------------------------------------
// Attention: flash-style, one wave per 16-query tile, 32-key blocks.
// 4 waves / 128 threads per block (1 wave per SIMD -> full VGPR budget).
// K fragments are software-pipelined across key blocks (double buffered),
// so their load latency overlaps the softmax + P*V work of the previous
// block instead of stalling the S WMMAs.
// q,k : [B,H,SEQ,HDIM] bf16 ; vT : [B,H,HDIM,SEQ] bf16 ;
// out : [T, D_MODEL] bf16 (A operand of the Wo projection GEMM).
// ---------------------------------------------------------------------------
#define S_STRIDE 36   // floats per S row in LDS (144 B: float4-aligned)

__global__ void __launch_bounds__(128, 1)
attention_wmma(const bf16_t* __restrict__ q,
               const bf16_t* __restrict__ k,
               const bf16_t* __restrict__ vT,
               bf16_t* __restrict__ attn_out) {
    __shared__ __align__(16) float Sl[4][16 * S_STRIDE]; // per-wave S tile
    __shared__ __align__(16) float Al[4][16];            // per-wave alpha/linv

    const int wave = threadIdx.x >> 5;
    const int lane = threadIdx.x & 31;
    const int ln15 = lane & 15;
    const int half = lane >> 4;

    const int bh = blockIdx.x >> 4;                        // 0..127  (b*8+h)
    const int q0 = ((blockIdx.x & 15) * 4 + wave) * 16;    // 0..1008
    const int b  = bh >> 3;
    const int h  = bh & 7;

    const bf16_t* qb = q  + (size_t)bh * SEQ  * HDIM;
    const bf16_t* kb = k  + (size_t)bh * SEQ  * HDIM;
    const bf16_t* vb = vT + (size_t)bh * HDIM * SEQ;

    // Q fragments for this 16-row tile (K = 96 -> 3 chunks of 32)
    v16bf qf[3];
#pragma unroll
    for (int c = 0; c < 3; ++c) {
        const bf16_t* p = qb + (size_t)(q0 + ln15) * HDIM + c * 32;
        qf[c] = mk16(*(const v8bf*)(p + 8 * half),
                     *(const v8bf*)(p + 16 + 8 * half));
    }

    auto loadK = [&](v16bf* f0, v16bf* f1, int j) {
#pragma unroll
        for (int c = 0; c < 3; ++c) {
            const bf16_t* p0 = kb + (size_t)(j + ln15) * HDIM + c * 32;
            f0[c] = mk16(*(const v8bf*)(p0 + 8 * half),
                         *(const v8bf*)(p0 + 16 + 8 * half));
            const bf16_t* p1 = kb + (size_t)(j + 16 + ln15) * HDIM + c * 32;
            f1[c] = mk16(*(const v8bf*)(p1 + 8 * half),
                         *(const v8bf*)(p1 + 16 + 8 * half));
        }
    };

    v8f o[6] = {v8f{}, v8f{}, v8f{}, v8f{}, v8f{}, v8f{}};
    float mrun = -1e30f;   // running row-max   (row = ln15, both halves agree)
    float lrun = 0.f;      // running row-sum
    const float scale = 0.1020620726f;   // 1/sqrt(96)

    float* Sw = &Sl[wave][0];
    float* Aw = &Al[wave][0];

    // prologue: preload K fragments for the first key block
    v16bf kc0[3], kc1[3], kn0[3], kn1[3];
    loadK(kc0, kc1, 0);

#pragma unroll 2
    for (int j0 = 0; j0 < SEQ; j0 += 32) {
        // ---- prefetch next key block (consumed next iteration) -----------
        const int jn = (j0 + 32 < SEQ) ? (j0 + 32) : 0;  // wrapped, stays in bounds
        loadK(kn0, kn1, jn);

        // ---- S = Q K^T for keys [j0, j0+32) using the preloaded buffer ---
        v8f s0 = v8f{}, s1 = v8f{};
#pragma unroll
        for (int c = 0; c < 3; ++c) s0 = wmma_bf16(qf[c], kc0[c], s0);
#pragma unroll
        for (int c = 0; c < 3; ++c) s1 = wmma_bf16(qf[c], kc1[c], s1);

        // ---- C->A transpose of S through LDS, storing (c, c+16) pairs ----
        // writer: element r lives at row (r + 8*half), col ln15
#pragma unroll
        for (int r = 0; r < 8; ++r) {
            float2* wp = (float2*)&Sw[(r + 8 * half) * S_STRIDE] + ln15;
            *wp = make_float2(s0[r] * scale, s1[r] * scale);
        }
        asm volatile("s_wait_dscnt 0" ::: "memory");

        // reader: lane's A-fragment slice = row ln15, col pairs 8*half..+7
        const float* srow = &Sw[ln15 * S_STRIDE + 16 * half];
        float arr[16];
#pragma unroll
        for (int i = 0; i < 4; ++i) {
            float4 f = ((const float4*)srow)[i];
            arr[4 * i + 0] = f.x; arr[4 * i + 1] = f.y;
            arr[4 * i + 2] = f.z; arr[4 * i + 3] = f.w;
        }

        // ---- online softmax (row = ln15; partner half-lane has the rest) --
        float lm = arr[0];
#pragma unroll
        for (int i = 1; i < 16; ++i) lm = fmaxf(lm, arr[i]);
        lm = fmaxf(lm, __shfl_xor(lm, 16, 32));
        const float mnew  = fmaxf(mrun, lm);
        const float alpha = __expf(mrun - mnew);
        float p[16], ls = 0.f;
#pragma unroll
        for (int i = 0; i < 16; ++i) { p[i] = __expf(arr[i] - mnew); ls += p[i]; }
        ls += __shfl_xor(ls, 16, 32);
        lrun = lrun * alpha + ls;
        mrun = mnew;

        // P is already in A-fragment order: pf[j] = col 8h+j, pf[8+j] = col 16+8h+j
        v16bf pf;
#pragma unroll
        for (int i = 0; i < 8; ++i) {
            pf[i]     = (bf16_t)p[2 * i];
            pf[i + 8] = (bf16_t)p[2 * i + 1];
        }

        // ---- broadcast alpha to C-layout lanes and rescale O -------------
        if (half == 0) Aw[ln15] = alpha;
        asm volatile("s_wait_dscnt 0" ::: "memory");
        float4 a0 = *(const float4*)(Aw + 8 * half);
        float4 a1 = *(const float4*)(Aw + 8 * half + 4);
        const float al[8] = {a0.x, a0.y, a0.z, a0.w, a1.x, a1.y, a1.z, a1.w};
#pragma unroll
        for (int t = 0; t < 6; ++t)
#pragma unroll
            for (int r = 0; r < 8; ++r) o[t][r] *= al[r];

        // ---- O += P(16x32) * V(32x96): load all 6 V fragments, then WMMAs -
        v16bf vf[6];
#pragma unroll
        for (int t = 0; t < 6; ++t) {
            const bf16_t* vp = vb + (size_t)(t * 16 + ln15) * SEQ + j0;
            vf[t] = mk16(*(const v8bf*)(vp + 8 * half),
                         *(const v8bf*)(vp + 16 + 8 * half));
        }
#pragma unroll
        for (int t = 0; t < 6; ++t)
            o[t] = wmma_bf16(pf, vf[t], o[t]);

        // ---- rotate K double-buffer (pure renaming under unroll-2) -------
#pragma unroll
        for (int c = 0; c < 3; ++c) { kc0[c] = kn0[c]; kc1[c] = kn1[c]; }
    }

    // ---- finalize: broadcast 1/l, normalize, scatter to [T, D_MODEL] -----
    if (half == 0) Aw[ln15] = 1.0f / lrun;
    asm volatile("s_wait_dscnt 0" ::: "memory");
    float4 a0 = *(const float4*)(Aw + 8 * half);
    float4 a1 = *(const float4*)(Aw + 8 * half + 4);
    const float inv[8] = {a0.x, a0.y, a0.z, a0.w, a1.x, a1.y, a1.z, a1.w};
#pragma unroll
    for (int r = 0; r < 8; ++r) {
        const int tok = q0 + r + 8 * half;
        const size_t base = ((size_t)b * SEQ + tok) * D_MODEL + h * HDIM;
#pragma unroll
        for (int t = 0; t < 6; ++t)
            attn_out[base + t * 16 + ln15] = (bf16_t)(o[t][r] * inv[r]);
    }
}

// ---------------------------------------------------------------------------
// Host orchestration
// ---------------------------------------------------------------------------
extern "C" void kernel_launch(void* const* d_in, const int* in_sizes, int n_in,
                              void* d_out, int out_size, void* d_ws, size_t ws_size,
                              hipStream_t stream) {
    (void)in_sizes; (void)n_in; (void)out_size; (void)ws_size;

    const float* x     = (const float*)d_in[0];
    const float* gamma = (const float*)d_in[1];
    const float* beta  = (const float*)d_in[2];
    const float* Wq = (const float*)d_in[3];  const float* bq = (const float*)d_in[4];
    const float* Wk = (const float*)d_in[5];  const float* bk = (const float*)d_in[6];
    const float* Wv = (const float*)d_in[7];  const float* bv = (const float*)d_in[8];
    const float* Wo = (const float*)d_in[9];  const float* bo = (const float*)d_in[10];
    const float* W1 = (const float*)d_in[11]; const float* b1 = (const float*)d_in[12];
    const float* W2 = (const float*)d_in[13]; const float* b2 = (const float*)d_in[14];

    char* ws = (char*)d_ws;
    size_t off = 0;
    auto alloc = [&](size_t bytes) -> char* {
        char* p = ws + off;
        off = (off + bytes + 255) & ~(size_t)255;
        return p;
    };

    bf16_t* WqT = (bf16_t*)alloc((size_t)D_MODEL * D_MODEL * 2);
    bf16_t* WkT = (bf16_t*)alloc((size_t)D_MODEL * D_MODEL * 2);
    bf16_t* WvT = (bf16_t*)alloc((size_t)D_MODEL * D_MODEL * 2);
    bf16_t* WoT = (bf16_t*)alloc((size_t)D_MODEL * D_MODEL * 2);
    bf16_t* W1T = (bf16_t*)alloc((size_t)D_MODEL * FFDIM * 2);
    bf16_t* W2T = (bf16_t*)alloc((size_t)FFDIM * D_MODEL * 2);
    bf16_t* hbf = (bf16_t*)alloc((size_t)TOKENS * D_MODEL * 2);   // LN1 out; reused as attn_out
    bf16_t* qb  = (bf16_t*)alloc((size_t)TOKENS * D_MODEL * 2);   // reused as h2 (LN2 out)
    bf16_t* kb  = (bf16_t*)alloc((size_t)TOKENS * D_MODEL * 2);
    bf16_t* vb  = (bf16_t*)alloc((size_t)TOKENS * D_MODEL * 2);
    float*  out1= (float* )alloc((size_t)TOKENS * D_MODEL * 4);   // attn residual sum
    bf16_t* gbf = (bf16_t*)alloc((size_t)TOKENS * FFDIM   * 2);   // GELU activations

    bf16_t* attn = hbf;   // alias: hbf dead after QKV GEMMs
    bf16_t* h2   = qb;    // alias: q dead after attention

    const dim3 tb(32, 8);
    // Weight transposes: in [K][N] -> out [N][K]
    transpose_to_bf16<<<dim3(D_MODEL/32, D_MODEL/32), tb, 0, stream>>>(Wq, WqT, D_MODEL, D_MODEL);
    transpose_to_bf16<<<dim3(D_MODEL/32, D_MODEL/32), tb, 0, stream>>>(Wk, WkT, D_MODEL, D_MODEL);
    transpose_to_bf16<<<dim3(D_MODEL/32, D_MODEL/32), tb, 0, stream>>>(Wv, WvT, D_MODEL, D_MODEL);
    transpose_to_bf16<<<dim3(D_MODEL/32, D_MODEL/32), tb, 0, stream>>>(Wo, WoT, D_MODEL, D_MODEL);
    transpose_to_bf16<<<dim3(FFDIM/32,   D_MODEL/32), tb, 0, stream>>>(W1, W1T, D_MODEL, FFDIM);
    transpose_to_bf16<<<dim3(D_MODEL/32, FFDIM/32),   tb, 0, stream>>>(W2, W2T, FFDIM, D_MODEL);

    // LN1
    layernorm_bf16<<<TOKENS, 256, 0, stream>>>(x, gamma, beta, hbf);

    // QKV projections (MODE 0: q/k layout, MODE 1: v-transposed layout)
    const dim3 gemmD(TOKENS / 128, D_MODEL / 64);
    gemm_wmma<0><<<gemmD, 256, 0, stream>>>(hbf, WqT, bq, nullptr, nullptr, qb,
                                            TOKENS, D_MODEL, D_MODEL);
    gemm_wmma<0><<<gemmD, 256, 0, stream>>>(hbf, WkT, bk, nullptr, nullptr, kb,
                                            TOKENS, D_MODEL, D_MODEL);
    gemm_wmma<1><<<gemmD, 256, 0, stream>>>(hbf, WvT, bv, nullptr, nullptr, vb,
                                            TOKENS, D_MODEL, D_MODEL);

    // Attention: 16*8 heads * 64 q-tiles / 4 waves-per-block = 2048 blocks
    attention_wmma<<<2048, 128, 0, stream>>>(qb, kb, vb, attn);

    // Output projection + residual(x) -> out1 (f32)
    gemm_wmma<2><<<gemmD, 256, 0, stream>>>(attn, WoT, bo, x, out1, nullptr,
                                            TOKENS, D_MODEL, D_MODEL);

    // LN2
    layernorm_bf16<<<TOKENS, 256, 0, stream>>>(out1, gamma, beta, h2);

    // FFN: GELU(h2 @ W1 + b1) -> gbf ; gbf @ W2 + b2 + out1 -> d_out
    gemm_wmma<3><<<dim3(TOKENS / 128, FFDIM / 64), 256, 0, stream>>>(
        h2, W1T, b1, nullptr, nullptr, gbf, TOKENS, FFDIM, D_MODEL);
    gemm_wmma<2><<<gemmD, 256, 0, stream>>>(gbf, W2T, b2, out1, (float*)d_out,
                                            nullptr, TOKENS, D_MODEL, FFDIM);
}
